// RNN_62259845923023
// MI455X (gfx1250) — matrix operand order repaired and checked by reading, not compiled, not run on hardware
//
#include <hip/hip_runtime.h>

// Elman RNN forward on gfx1250:
//   h_t = tanh([x_t, h_{t-1}] @ Wc + (b1+b2)),  Wc = [W1^T; W2^T]  (192 x 128)
// 32 workgroups (one 16-row batch tile each), 256 threads = 8 wave32,
// wave w owns output columns [16w, 16w+16). Weights live in VGPRs as WMMA
// B-fragments; recurrence state ping-pongs between two LDS buffers.
// Critical path per step: 12x ds_load_b128 -> 2 interleaved WMMA chains ->
// v_tanh_f32 -> LDS publish -> ONE barrier. x_{t+1} global load issued at the
// top of the step so its latency hides under the WMMA chain; h_seq global
// stores are deferred one step and drain against STOREcnt in the background.

#define S_LEN   2048
#define BATCHN  512
#define ISZ     64
#define HSZ     128
#define KTOT    192          // ISZ + HSZ
#define LDA     216          // f16 elems/row: 432B = 16B-aligned, conflict-free b128
#define NCHUNK  6            // KTOT / 32

typedef __attribute__((ext_vector_type(16))) _Float16 v16h;
typedef __attribute__((ext_vector_type(8)))  _Float16 v8h;
typedef __attribute__((ext_vector_type(4)))  _Float16 v4h;
typedef __attribute__((ext_vector_type(8)))  float    v8f;

// LDS-only workgroup sync: waits DScnt but NOT STOREcnt, so global h_seq
// stores never gate the recurrence.
static __device__ __forceinline__ void wg_sync_lds() {
#if __has_builtin(__builtin_amdgcn_s_barrier_signal) && \
    __has_builtin(__builtin_amdgcn_s_barrier_wait)
    asm volatile("s_wait_dscnt 0x0" ::: "memory");
    __builtin_amdgcn_s_barrier_signal(-1);
    __builtin_amdgcn_s_barrier_wait(-1);
#else
    __syncthreads();
#endif
}

static __device__ __forceinline__ float fast_tanh(float s) {
#if __has_builtin(__builtin_amdgcn_tanhf)
    return __builtin_amdgcn_tanhf(s);          // v_tanh_f32
#else
    const float z = s * 2.885390081777927f;    // 2 * log2(e)
    const float e = __builtin_amdgcn_exp2f(z);
    const float r = __builtin_amdgcn_rcpf(e + 1.0f);
    return __builtin_fmaf(-2.0f, r, 1.0f);
#endif
}

__global__ __launch_bounds__(256, 1)
void rnn_scan_wmma(const float* __restrict__ x,
                   const float* __restrict__ W1,
                   const float* __restrict__ b1,
                   const float* __restrict__ W2,
                   const float* __restrict__ b2,
                   float* __restrict__ out)
{
    __shared__ _Float16 Alds[2][16 * LDA];  // ping-pong [x_t | h] tiles (f16)
    __shared__ float    Stg [2][16 * HSZ];  // ping-pong f32 h staging

    const int tid  = threadIdx.x;
    const int lane = tid & 31;
    const int wv   = tid >> 5;          // 0..7 : N-tile index
    const int g    = lane >> 4;         // lane half (K-group select)
    const int l16  = lane & 15;
    const int b0   = blockIdx.x * 16;   // batch row base

    // ---- preload combined-weight B fragments (once) ----
    // B layout (K x N = 32 x 16): lanes 0-15 hold K=0..15, lanes 16-31 K=16..31,
    // N = lane%16, element j of v16h = K = 32c + 16g + j.
    const int ncol = wv * 16 + l16;
    v16h Bf[NCHUNK];
#pragma unroll
    for (int c = 0; c < NCHUNK; ++c) {
#pragma unroll
        for (int j = 0; j < 16; ++j) {
            const int k = c * 32 + g * 16 + j;
            const float w = (k < ISZ) ? W1[ncol * ISZ + k]
                                      : W2[ncol * HSZ + (k - ISZ)];
            Bf[c][j] = (_Float16)w;
        }
    }
    const float bsum = b1[ncol] + b2[ncol];

    const int xrow = tid >> 4;           // 16 threads/row * 4 f32 = 64 cols
    const int xcol = (tid & 15) * 4;

    // ---- init buffer 0: h_0 = 0, stage x_0 ----
    for (int i = tid; i < 16 * HSZ; i += 256)
        Alds[0][(i >> 7) * LDA + ISZ + (i & 127)] = (_Float16)0.f;
    {
        const float4 xv = *(const float4*)&x[((size_t)0 * BATCHN + b0 + xrow) * ISZ + xcol];
        v4h h4 = { (_Float16)xv.x, (_Float16)xv.y, (_Float16)xv.z, (_Float16)xv.w };
        *(v4h*)&Alds[0][xrow * LDA + xcol] = h4;
    }
    __syncthreads();

    // One RNN step: read Ard/Srd, write Awr/Swr, single barrier at the end.
    auto step = [&](int t, const _Float16* Ard, _Float16* Awr,
                    const float* Srd, float* Swr) __attribute__((always_inline)) {
        // Issue x_{t+1} load FIRST: latency hides under the full WMMA chain.
        float4 xv;
        const bool havex = (t + 1) < S_LEN;
        if (havex)
            xv = *(const float4*)&x[((size_t)(t + 1) * BATCHN + b0 + xrow) * ISZ + xcol];
        if (t + 4 < S_LEN)
            __builtin_prefetch(&x[((size_t)(t + 4) * BATCHN + b0 + xrow) * ISZ + xcol], 0, 3);

        // ---- A-fragment loads (12x ds_load_b128, distinct regs) ----
        v8h lo[NCHUNK], hi[NCHUNK];
#pragma unroll
        for (int c = 0; c < NCHUNK; ++c) {
            const int kb = c * 32 + g * 16;
            lo[c] = *(const v8h*)&Ard[l16 * LDA + kb];
            hi[c] = *(const v8h*)&Ard[l16 * LDA + kb + 8];
        }

        // ---- two independent WMMA chains ----
        v8f acc0, acc1;
#pragma unroll
        for (int r = 0; r < 8; ++r) { acc0[r] = bsum; acc1[r] = 0.0f; }
#pragma unroll
        for (int c = 0; c < NCHUNK; c += 2) {
            v16h a0, a1;
#pragma unroll
            for (int i = 0; i < 8; ++i) {
                a0[i] = lo[c][i];     a0[8 + i] = hi[c][i];
                a1[i] = lo[c + 1][i]; a1[8 + i] = hi[c + 1][i];
            }
            acc0 = __builtin_amdgcn_wmma_f32_16x16x32_f16(
                       false, a0, false, Bf[c],     (short)0, acc0, false, false);
            acc1 = __builtin_amdgcn_wmma_f32_16x16x32_f16(
                       false, a1, false, Bf[c + 1], (short)0, acc1, false, false);
        }

        float hv[8];
#pragma unroll
        for (int r = 0; r < 8; ++r)
            hv[r] = fast_tanh(acc0[r] + acc1[r]);   // v_tanh_f32

        // ---- publish h into the OTHER buffer (no barrier needed before) ----
        // D layout: VGPR r -> row M = r + 8g, col N = l16.
#pragma unroll
        for (int r = 0; r < 8; ++r) {
            const int M = r + 8 * g;
            Awr[M * LDA + ISZ + ncol] = (_Float16)hv[r];
            Swr[M * HSZ + ncol]       = hv[r];
        }
        if (havex) {
            v4h h4 = { (_Float16)xv.x, (_Float16)xv.y, (_Float16)xv.z, (_Float16)xv.w };
            *(v4h*)&Awr[xrow * LDA + xcol] = h4;
        }

        // ---- deferred coalesced store of h_{t-1} (fire-and-forget) ----
        if (t > 0) {
#pragma unroll
            for (int j = 0; j < 2; ++j) {
                const int e   = tid * 4 + j * 1024;
                const int row = e >> 7, col = e & 127;
                const float4 v = *(const float4*)&Srd[row * HSZ + col];
                *(float4*)&out[((size_t)(t - 1) * BATCHN + b0 + row) * HSZ + col] = v;
            }
        }

        wg_sync_lds();   // the ONLY barrier per step
    };

    // Unroll by 2 so ping-pong buffer indices are compile-time constants.
    for (int t = 0; t < S_LEN; t += 2) {
        step(t,     Alds[0], Alds[1], Stg[1], Stg[0]);
        step(t + 1, Alds[1], Alds[0], Stg[0], Stg[1]);
    }

    // ---- h_{S-1}: stored once as the last h_seq slice and once as h_last ----
    // (last written buffer is Stg[(S_LEN-1) & 1] = Stg[1])
#pragma unroll
    for (int j = 0; j < 2; ++j) {
        const int e   = tid * 4 + j * 1024;
        const int row = e >> 7, col = e & 127;
        const float4 v = *(const float4*)&Stg[1][row * HSZ + col];
        *(float4*)&out[((size_t)(S_LEN - 1) * BATCHN + b0 + row) * HSZ + col] = v;
        *(float4*)&out[(size_t)S_LEN * BATCHN * HSZ + (size_t)(b0 + row) * HSZ + col] = v;
    }
}

extern "C" void kernel_launch(void* const* d_in, const int* in_sizes, int n_in,
                              void* d_out, int out_size, void* d_ws, size_t ws_size,
                              hipStream_t stream)
{
    (void)in_sizes; (void)n_in; (void)out_size; (void)d_ws; (void)ws_size;
    const float* x  = (const float*)d_in[0];
    const float* W1 = (const float*)d_in[1];
    const float* b1 = (const float*)d_in[2];
    const float* W2 = (const float*)d_in[3];
    const float* b2 = (const float*)d_in[4];
    float* out = (float*)d_out;

    dim3 grid(BATCHN / 16);   // 32 workgroups
    dim3 block(256);          // 8 wave32
    rnn_scan_wmma<<<grid, block, 0, stream>>>(x, W1, b1, W2, b2, out);
}
